// Decoder_52656299049337
// MI455X (gfx1250) — compile-verified, matrix operand order checked by reading, not run on hardware
//
#include <hip/hip_runtime.h>
#include <hip/hip_bf16.h>

// ---------------------------------------------------------------------------
// TreeGCN point-cloud decoder for MI455X (gfx1250, wave32, WMMA).
//
// Pipeline (B=2):
//   knn(512) -> op3d -> gemm(feat@w1, WMMA) -> opnd -> treeGCN -> pc1(2,4096,3)
//   knn(4096) -> op3d -> [gemm+opnd] x3 -> treeGCN -> pc2(2,32768,3)
//
// Key optimizations:
//   * loopA @ loopB collapsed to a 64x3 matrix (matmul associativity):
//     removes ~2.7 GMAC + a 168MB intermediate from the hot path.
//   * branch weights (536 MB) streamed exactly once; both batch elements
//     computed in the same block (memory-bound: ~23us floor at 23.3 TB/s).
//   * knn fused with top-k and chunked 4-way over candidates for wave32
//     occupancy; no NxN distance matrix is ever materialized.
//   * feat@w GEMMs use v_wmma_f32_16x16x32_f16; W is pre-packed to f16
//     fragment layout so B loads are 2x global_load_b128, A loads float4.
// ---------------------------------------------------------------------------

typedef __attribute__((ext_vector_type(16))) _Float16 v16h;
typedef __attribute__((ext_vector_type(8)))  float    v8f;

__device__ __forceinline__ float lrelu(float v) { return v >= 0.f ? v : 0.2f * v; }

// ---------------------------------------------------------------------------
// Column-wise direction normalization: src,dst are (3, C) row-major.
// ---------------------------------------------------------------------------
__global__ void norm_cols_kernel(const float* __restrict__ src,
                                 float* __restrict__ dst, int C) {
  int c = blockIdx.x * blockDim.x + threadIdx.x;
  if (c >= C) return;
  float a = src[c], b = src[C + c], d = src[2 * C + c];
  float inv = 1.f / fmaxf(sqrtf(a * a + b * b + d * d), 1e-12f);
  dst[c] = a * inv; dst[C + c] = b * inv; dst[2 * C + c] = d * inv;
}

// ---------------------------------------------------------------------------
// Wl = loopA(64x640) @ loopB(640x3)  -> (64x3). 192 threads, one block.
// ---------------------------------------------------------------------------
__global__ void matmul_small_kernel(const float* __restrict__ A,
                                    const float* __restrict__ B,
                                    float* __restrict__ out) {
  int t = threadIdx.x;
  if (t >= 192) return;
  int i = t / 3, j = t % 3;
  float s = 0.f;
  for (int k = 0; k < 640; ++k) s += A[i * 640 + k] * B[k * 3 + j];
  out[i * 3 + j] = s;
}

// ---------------------------------------------------------------------------
// kNN phase 1: per (row, chunk) top-20 over N/nchunks candidates.
// grid = (ceil(N/256), nchunks, B); dynamic LDS = 4*(N/nchunks) floats.
// Register-resident fully-unrolled insertion buffer. Downstream max/sum over
// k is order-invariant, so only set membership must match the reference;
// self is excluded (== reference top-21-drop-first for distinct points).
// ---------------------------------------------------------------------------
__global__ void knn_partial_kernel(const float* __restrict__ pts,
                                   float* __restrict__ pdist,
                                   int* __restrict__ pidx,
                                   int N, int nchunks) {
  extern __shared__ float sm[];
  int len = N / nchunks;
  float* sp = sm;            // len*3
  float* sq = sm + len * 3;  // len
  int b = blockIdx.z, chunk = blockIdx.y;
  int c0 = chunk * len;
  const float* P = pts + (long)b * N * 3;
  for (int i = threadIdx.x; i < len; i += blockDim.x) {
    int m = c0 + i;
    float x = P[m * 3], y = P[m * 3 + 1], z = P[m * 3 + 2];
    sp[i * 3] = x; sp[i * 3 + 1] = y; sp[i * 3 + 2] = z;
    sq[i] = x * x + y * y + z * z;
  }
  __syncthreads();
  int r = blockIdx.x * blockDim.x + threadIdx.x;
  if (r >= N) return;
  float rx = P[r * 3], ry = P[r * 3 + 1], rz = P[r * 3 + 2];
  float rq = rx * rx + ry * ry + rz * rz;
  float bd[20]; int bi[20];
#pragma unroll
  for (int j = 0; j < 20; ++j) { bd[j] = 3.0e38f; bi[j] = 0; }
  for (int i = 0; i < len; ++i) {
    int m = c0 + i;
    if (m == r) continue;
    float d = rq - 2.0f * (rx * sp[i * 3] + ry * sp[i * 3 + 1] + rz * sp[i * 3 + 2]) + sq[i];
    if (d < bd[19]) {
      float dd = d; int ii = m;
#pragma unroll
      for (int j = 0; j < 20; ++j) {
        if (dd < bd[j]) {
          float td = bd[j]; int ti = bi[j];
          bd[j] = dd; bi[j] = ii; dd = td; ii = ti;
        }
      }
    }
  }
  long base = (((long)b * N + r) * nchunks + chunk) * 20;
#pragma unroll
  for (int j = 0; j < 20; ++j) { pdist[base + j] = bd[j]; pidx[base + j] = bi[j]; }
}

// ---------------------------------------------------------------------------
// kNN phase 2: merge nchunks sorted 20-lists per row -> final 20 indices.
// Partial lists are ascending, so break out as soon as a list can't improve.
// ---------------------------------------------------------------------------
__global__ void knn_merge_kernel(const float* __restrict__ pdist,
                                 const int* __restrict__ pidx,
                                 int* __restrict__ idx,
                                 int rows, int nchunks) {
  int r = blockIdx.x * blockDim.x + threadIdx.x;
  if (r >= rows) return;
  float bd[20]; int bi[20];
#pragma unroll
  for (int j = 0; j < 20; ++j) { bd[j] = 3.0e38f; bi[j] = 0; }
  for (int c = 0; c < nchunks; ++c) {
    long base = ((long)r * nchunks + c) * 20;
    for (int j = 0; j < 20; ++j) {
      float d = pdist[base + j];
      if (d >= bd[19]) break;          // ascending partial list: done
      float dd = d; int ii = pidx[base + j];
#pragma unroll
      for (int q = 0; q < 20; ++q) {
        if (dd < bd[q]) {
          float td = bd[q]; int ti = bi[q];
          bd[q] = dd; bi[q] = ii; dd = td; ii = ti;
        }
      }
    }
  }
  int* op = idx + (long)r * 20;
#pragma unroll
  for (int j = 0; j < 20; ++j) op[j] = bi[j];
}

// ---------------------------------------------------------------------------
// op3d: out[p, c] = relu( sum_s max_k relu(ndn_k . sdn[:, s*32+c]) )
// One wave per point (lane = channel, 32 ch). Block = 8 points.
// ---------------------------------------------------------------------------
__global__ void op3d_kernel(const float* __restrict__ pts,
                            const int* __restrict__ idx,
                            const float* __restrict__ sdn,   // (3,320)
                            float* __restrict__ out, int N) {
  __shared__ float ndn[8][20][3];
  int g = threadIdx.x >> 5, c = threadIdx.x & 31;
  int p = blockIdx.x * 8 + g;
  int b = p / N, n = p % N;
  const float* P = pts + (long)b * N * 3;
  if (c < 20) {
    int id = idx[(long)p * 20 + c];
    float dx = P[id * 3]     - P[n * 3];
    float dy = P[id * 3 + 1] - P[n * 3 + 1];
    float dz = P[id * 3 + 2] - P[n * 3 + 2];
    float inv = 1.f / fmaxf(sqrtf(dx * dx + dy * dy + dz * dz), 1e-12f);
    ndn[g][c][0] = dx * inv; ndn[g][c][1] = dy * inv; ndn[g][c][2] = dz * inv;
  }
  __syncthreads();
  float acc = 0.f;
  for (int s = 0; s < 10; ++s) {
    int e = s * 32 + c;
    float d0 = sdn[e], d1 = sdn[320 + e], d2 = sdn[640 + e];
    float mx = 0.f;  // relu(theta) >= 0, so 0 init == max over k
#pragma unroll
    for (int k = 0; k < 20; ++k) {
      float t = ndn[g][k][0] * d0 + ndn[g][k][1] * d1 + ndn[g][k][2] * d2;
      mx = fmaxf(mx, fmaxf(t, 0.f));
    }
    acc += mx;
  }
  out[(long)p * 32 + c] = fmaxf(acc, 0.f);
}

// ---------------------------------------------------------------------------
// Pre-pack W (KxN f32, row-major) into f16 WMMA B-fragment blocks:
// block (kb, tn) = 512 halves; lane l owns 32 bytes at blockBase + l*32,
// element i = W[kb + kh + (i<8 ? i : 8+i)][tn*16 + lid]  (ISA 7.12.2 layout:
// lane<16 -> K{0..7,16..23}, lane>=16 -> K{8..15,24..31} of column tn*16+lid).
// One thread per (kb, tn, lane).
// ---------------------------------------------------------------------------
__global__ void pack_w_kernel(const float* __restrict__ W,
                              _Float16* __restrict__ Wh, int K, int N) {
  int tilesN = N / 16;
  int total = (K / 32) * tilesN * 32;
  int tid = blockIdx.x * blockDim.x + threadIdx.x;
  if (tid >= total) return;
  int lane = tid & 31;
  int tmp  = tid >> 5;
  int tn   = tmp % tilesN;
  int kb   = (tmp / tilesN) * 32;
  int hsel = lane >> 4, lid = lane & 15;
  int col  = tn * 16 + lid;
  int kh   = kb + hsel * 8;
  _Float16* dst = Wh + (long)tid * 16;
#pragma unroll
  for (int i = 0; i < 8; ++i) {
    dst[i]     = (_Float16)W[(long)(kh + i) * N + col];
    dst[i + 8] = (_Float16)W[(long)(kh + 16 + i) * N + col];
  }
}

// ---------------------------------------------------------------------------
// WMMA GEMM: out(MxN) = A(MxK,f32) @ W(KxN) + bias, f16 in / f32 accumulate.
// One 16x16 tile per wave; K in {32,64}; N=704=44*16; tile guard is
// wave-uniform so EXEC is all-ones at every WMMA (ISA 7.12 requirement).
// A loaded as float4 pairs (rows 32B-aligned), B as one aligned v16h
// (pre-packed fragment layout -> 2x global_load_b128 per K-step).
// ---------------------------------------------------------------------------
__global__ void gemm_wmma_kernel(const float* __restrict__ A,
                                 const _Float16* __restrict__ Wh,
                                 const float* __restrict__ bias,
                                 float* __restrict__ out,
                                 int K, int N, int tilesN, int numTiles) {
  int lane = threadIdx.x & 31;
  int wid  = threadIdx.x >> 5;
  int tile = blockIdx.x * (blockDim.x >> 5) + wid;
  if (tile >= numTiles) return;           // wave-uniform exit
  int tm = tile / tilesN, tn = tile % tilesN;
  int hsel = lane >> 4;                   // 0 or 1
  int lid  = lane & 15;
  int row  = tm * 16 + lid;
  int col  = tn * 16 + lid;
  int kh   = hsel * 8;
  const v16h* Bfrag = reinterpret_cast<const v16h*>(Wh);
  v8f acc = {};
  for (int kb = 0; kb < K; kb += 32) {
    const float4* ap4 =
        reinterpret_cast<const float4*>(A + (long)row * K + kb + kh);
    float4 x0 = ap4[0];   // K = kh+0..3
    float4 x1 = ap4[1];   // K = kh+4..7
    float4 x2 = ap4[4];   // K = kh+16..19
    float4 x3 = ap4[5];   // K = kh+20..23
    v16h a;
    a[0]  = (_Float16)x0.x; a[1]  = (_Float16)x0.y;
    a[2]  = (_Float16)x0.z; a[3]  = (_Float16)x0.w;
    a[4]  = (_Float16)x1.x; a[5]  = (_Float16)x1.y;
    a[6]  = (_Float16)x1.z; a[7]  = (_Float16)x1.w;
    a[8]  = (_Float16)x2.x; a[9]  = (_Float16)x2.y;
    a[10] = (_Float16)x2.z; a[11] = (_Float16)x2.w;
    a[12] = (_Float16)x3.x; a[13] = (_Float16)x3.y;
    a[14] = (_Float16)x3.z; a[15] = (_Float16)x3.w;
    v16h bm = Bfrag[((long)(kb >> 5) * tilesN + tn) * 32 + lane];
    acc = __builtin_amdgcn_wmma_f32_16x16x32_f16(
        false, a, false, bm, (short)0, acc, false, false);
  }
  float bv = bias[col];
  int rbase = tm * 16 + hsel * 8;
  float* op = out + (long)rbase * N + col;
#pragma unroll
  for (int r = 0; r < 8; ++r) op[(long)r * N] = acc[r] + bv;
}

// ---------------------------------------------------------------------------
// opnd aggregation: out[p,c] = relu( fo[p,c] +
//    sum_s max_k relu(ndn_k . sdn[:,s*64+c]) * fo[nb_k, 64 + s*64 + c] )
// 64 threads per point (c = channel), block = 4 points.
// ---------------------------------------------------------------------------
__global__ void opnd_kernel(const float* __restrict__ pts,
                            const int* __restrict__ idx,
                            const float* __restrict__ fo,    // (B*N, 704)
                            const float* __restrict__ sdn,   // (3, 640)
                            float* __restrict__ out, int N) {
  __shared__ float ndn[4][20][3];
  __shared__ int   nid[4][20];
  int g = threadIdx.x >> 6, c = threadIdx.x & 63;
  int p = blockIdx.x * 4 + g;
  int b = p / N, n = p % N;
  const float* P = pts + (long)b * N * 3;
  if (c < 20) {
    int id = idx[(long)p * 20 + c];
    nid[g][c] = id;
    float dx = P[id * 3]     - P[n * 3];
    float dy = P[id * 3 + 1] - P[n * 3 + 1];
    float dz = P[id * 3 + 2] - P[n * 3 + 2];
    float inv = 1.f / fmaxf(sqrtf(dx * dx + dy * dy + dz * dz), 1e-12f);
    ndn[g][c][0] = dx * inv; ndn[g][c][1] = dy * inv; ndn[g][c][2] = dz * inv;
  }
  __syncthreads();
  float fc = fo[(long)p * 704 + c];
  float acc = 0.f;
  for (int s = 0; s < 10; ++s) {
    int e = s * 64 + c;
    float d0 = sdn[e], d1 = sdn[640 + e], d2 = sdn[1280 + e];
    float mx = -3.0e38f;
#pragma unroll
    for (int k = 0; k < 20; ++k) {
      float t = fmaxf(ndn[g][k][0] * d0 + ndn[g][k][1] * d1 + ndn[g][k][2] * d2, 0.f);
      float fsv = fo[((long)b * N + nid[g][k]) * 704 + 64 + e];
      mx = fmaxf(mx, t * fsv);
    }
    acc += mx;
  }
  out[(long)p * 64 + c] = fmaxf(fc + acc, 0.f);
}

// ---------------------------------------------------------------------------
// TreeGCN: out[b, n*8+d, :] = root_b + sum_f lrelu(x_b . Wb[n,:,d*64+f]) * Wl[f,:]
// (+bias, lrelu if act). One block per node n; wave w owns d=w; each lane
// owns 2 branch features for both batch elements; shfl_xor wave reduction.
// Streams Wb[n] (128 KB) exactly once for both batches (536 MB total).
// ---------------------------------------------------------------------------
__global__ void tree_kernel(const float* __restrict__ x,      // (B, N, 64)
                            const float* __restrict__ Wb,     // (N, 64, 512)
                            const float* __restrict__ rootW,  // (64, 3)
                            const float* __restrict__ Wl,     // (64, 3)
                            const float* __restrict__ bias,   // (8, 3) or null
                            float* __restrict__ out,          // (B, N*8, 3)
                            int N, int act) {
  __shared__ float xs[2][64];
  __shared__ float root[2][3];
  int n = blockIdx.x, t = threadIdx.x;
  if (t < 128) {
    int b = t >> 6, i = t & 63;
    xs[b][i] = x[((long)b * N + n) * 64 + i];
  }
  __syncthreads();
  if (t < 6) {
    int b = t / 3, j = t % 3;
    float s = 0.f;
    for (int i = 0; i < 64; ++i) s += xs[b][i] * rootW[i * 3 + j];
    root[b][j] = s;
  }
  __syncthreads();
  int w = t >> 5, l = t & 31;       // w = degree slot d, two features per lane
  int f0 = 2 * l, f1 = 2 * l + 1;
  const float2* wp2 = reinterpret_cast<const float2*>(
      Wb + (long)n * 64 * 512 + w * 64 + f0);
  float d00 = 0.f, d01 = 0.f, d10 = 0.f, d11 = 0.f;
#pragma unroll 4
  for (int i = 0; i < 64; ++i) {
    float2 wv = wp2[i * 256];
    float x0 = xs[0][i], x1 = xs[1][i];
    d00 += x0 * wv.x; d01 += x0 * wv.y;
    d10 += x1 * wv.x; d11 += x1 * wv.y;
  }
  float h00 = lrelu(d00), h01 = lrelu(d01), h10 = lrelu(d10), h11 = lrelu(d11);
  float a0[3], a1[3];
#pragma unroll
  for (int j = 0; j < 3; ++j) {
    float wl0 = Wl[f0 * 3 + j], wl1 = Wl[f1 * 3 + j];
    a0[j] = h00 * wl0 + h01 * wl1;
    a1[j] = h10 * wl0 + h11 * wl1;
  }
#pragma unroll
  for (int off = 16; off > 0; off >>= 1) {
#pragma unroll
    for (int j = 0; j < 3; ++j) {
      a0[j] += __shfl_xor(a0[j], off, 32);
      a1[j] += __shfl_xor(a1[j], off, 32);
    }
  }
  if (l == 0) {
    long o0 = ((long)n * 8 + w) * 3;
    long o1 = ((long)N * 8 + (long)n * 8 + w) * 3;
#pragma unroll
    for (int j = 0; j < 3; ++j) {
      float v0 = root[0][j] + a0[j];
      float v1 = root[1][j] + a1[j];
      if (act) {
        float bv = bias[w * 3 + j];
        v0 = lrelu(v0 + bv);
        v1 = lrelu(v1 + bv);
      }
      out[o0 + j] = v0;
      out[o1 + j] = v1;
    }
  }
}

// ---------------------------------------------------------------------------
// Host orchestration.
// ---------------------------------------------------------------------------
extern "C" void kernel_launch(void* const* d_in, const int* in_sizes, int n_in,
                              void* d_out, int out_size, void* d_ws, size_t ws_size,
                              hipStream_t stream) {
  (void)in_sizes; (void)n_in; (void)out_size; (void)ws_size;
  const float* coase   = (const float*)d_in[0];   // (2, 512, 3)
  const float* d0      = (const float*)d_in[1];   // (3, 320)
  const float* w1      = (const float*)d_in[2];   // (32, 704)
  const float* b1      = (const float*)d_in[3];   // (704)
  const float* dir1    = (const float*)d_in[4];   // (3, 640)
  const float* d2      = (const float*)d_in[5];   // (3, 320)
  const float* w3      = (const float*)d_in[6];   // (32, 704)
  const float* b3      = (const float*)d_in[7];
  const float* dir3    = (const float*)d_in[8];
  const float* w4      = (const float*)d_in[9];   // (64, 704)
  const float* b4      = (const float*)d_in[10];
  const float* dir4    = (const float*)d_in[11];
  const float* w5      = (const float*)d_in[12];  // (64, 704)
  const float* b5      = (const float*)d_in[13];
  const float* dir5    = (const float*)d_in[14];
  const float* root1   = (const float*)d_in[15];  // (64, 3)
  const float* branch1 = (const float*)d_in[16];  // (512, 64, 512)
  const float* loop1a  = (const float*)d_in[17];  // (64, 640)
  const float* loop1b  = (const float*)d_in[18];  // (640, 3)
  const float* bias1   = (const float*)d_in[19];  // (1, 8, 3)
  const float* root2   = (const float*)d_in[20];  // (64, 3)
  const float* branch2 = (const float*)d_in[21];  // (4096, 64, 512)
  const float* loop2a  = (const float*)d_in[22];  // (64, 640)
  const float* loop2b  = (const float*)d_in[23];  // (640, 3)

  float* pc1 = (float*)d_out;          // (2, 4096, 3) = 24576 floats
  float* pc2 = pc1 + 2 * 4096 * 3;     // (2, 32768, 3) = 196608 floats

  // Workspace layout (32B-aligned float regions), ~37 MB total.
  float* ws = (float*)d_ws;
  size_t off = 0;
  auto alloc = [&](size_t nf) {
    off = (off + 7) & ~(size_t)7;
    float* p = ws + off; off += nf; return p;
  };
  float* sdn0 = alloc(3 * 320);
  float* sdn1 = alloc(3 * 640);
  float* sdn2 = alloc(3 * 320);
  float* sdn3 = alloc(3 * 640);
  float* sdn4 = alloc(3 * 640);
  float* sdn5 = alloc(3 * 640);
  float* Wl1  = alloc(64 * 3);
  float* Wl2  = alloc(64 * 3);
  float* fm0  = alloc(1024 * 32);
  float* fm1  = alloc(1024 * 64);
  float* fm2  = alloc(8192 * 32);
  float* fm3  = alloc(8192 * 64);
  float* fm4  = alloc(8192 * 64);
  float* fm5  = alloc(8192 * 64);
  float* fo   = alloc((size_t)8192 * 704);
  _Float16* Wh = (_Float16*)alloc(2 * 44 * 512);          // packed W, K<=64
  float* pdist = alloc((size_t)2 * 4096 * 4 * 20);        // knn partials
  int* pidx = (int*)alloc((size_t)2 * 4096 * 4 * 20);
  int* idx1 = (int*)alloc(2 * 512 * 20);
  int* idx2 = (int*)alloc(2 * 4096 * 20);

  // --- tiny preprocessing ---
  norm_cols_kernel<<<2, 256, 0, stream>>>(d0,   sdn0, 320);
  norm_cols_kernel<<<3, 256, 0, stream>>>(dir1, sdn1, 640);
  norm_cols_kernel<<<2, 256, 0, stream>>>(d2,   sdn2, 320);
  norm_cols_kernel<<<3, 256, 0, stream>>>(dir3, sdn3, 640);
  norm_cols_kernel<<<3, 256, 0, stream>>>(dir4, sdn4, 640);
  norm_cols_kernel<<<3, 256, 0, stream>>>(dir5, sdn5, 640);
  matmul_small_kernel<<<1, 192, 0, stream>>>(loop1a, loop1b, Wl1);
  matmul_small_kernel<<<1, 192, 0, stream>>>(loop2a, loop2b, Wl2);

  // --- stage 1: 512 points (2 candidate chunks of 256) ---
  knn_partial_kernel<<<dim3(2, 2, 2), 256, 4 * 256 * sizeof(float), stream>>>(
      coase, pdist, pidx, 512, 2);
  knn_merge_kernel<<<4, 256, 0, stream>>>(pdist, pidx, idx1, 1024, 2);
  op3d_kernel<<<128, 256, 0, stream>>>(coase, idx1, sdn0, fm0, 512);
  pack_w_kernel<<<6, 256, 0, stream>>>(w1, Wh, 32, 704);      // 1408 threads
  gemm_wmma_kernel<<<(2816 + 7) / 8, 256, 0, stream>>>(       // 64*44 tiles
      fm0, Wh, b1, fo, 32, 704, 44, 2816);
  opnd_kernel<<<256, 256, 0, stream>>>(coase, idx1, fo, sdn1, fm1, 512);
  tree_kernel<<<512, 256, 0, stream>>>(fm1, branch1, root1, Wl1, bias1, pc1, 512, 1);

  // --- stage 2: 4096 points (4 candidate chunks of 1024) ---
  knn_partial_kernel<<<dim3(16, 4, 2), 256, 4 * 1024 * sizeof(float), stream>>>(
      pc1, pdist, pidx, 4096, 4);
  knn_merge_kernel<<<32, 256, 0, stream>>>(pdist, pidx, idx2, 8192, 4);
  op3d_kernel<<<1024, 256, 0, stream>>>(pc1, idx2, sdn2, fm2, 4096);

  pack_w_kernel<<<6, 256, 0, stream>>>(w3, Wh, 32, 704);
  gemm_wmma_kernel<<<(22528 + 7) / 8, 256, 0, stream>>>(      // 512*44 tiles
      fm2, Wh, b3, fo, 32, 704, 44, 22528);
  opnd_kernel<<<2048, 256, 0, stream>>>(pc1, idx2, fo, sdn3, fm3, 4096);

  pack_w_kernel<<<11, 256, 0, stream>>>(w4, Wh, 64, 704);     // 2816 threads
  gemm_wmma_kernel<<<(22528 + 7) / 8, 256, 0, stream>>>(
      fm3, Wh, b4, fo, 64, 704, 44, 22528);
  opnd_kernel<<<2048, 256, 0, stream>>>(pc1, idx2, fo, sdn4, fm4, 4096);

  pack_w_kernel<<<11, 256, 0, stream>>>(w5, Wh, 64, 704);
  gemm_wmma_kernel<<<(22528 + 7) / 8, 256, 0, stream>>>(
      fm4, Wh, b5, fo, 64, 704, 44, 22528);
  opnd_kernel<<<2048, 256, 0, stream>>>(pc1, idx2, fo, sdn5, fm5, 4096);

  tree_kernel<<<4096, 256, 0, stream>>>(fm5, branch2, root2, Wl2, nullptr, pc2, 4096, 0);
}
